// GraphEncoder_12146167513685
// MI455X (gfx1250) — matrix operand order repaired and checked by reading, not compiled, not run on hardware
//
#include <hip/hip_runtime.h>
#include <hip/hip_bf16.h>
#include <math.h>

typedef __attribute__((ext_vector_type(16))) _Float16 v16h;
typedef __attribute__((ext_vector_type(8)))  _Float16 v8h;
typedef __attribute__((ext_vector_type(8)))  float    v8f;

namespace {
constexpr int kN    = 2048;
constexpr int kD    = 256;
constexpr int kH    = 8;
constexpr int kDK   = 32;
constexpr int kL    = 4;
constexpr int kDFF  = 1024;
}

// ---------------- WMMA fragment helpers (wave32, CDNA5 layouts) -------------
// A fragment (16x32 f16) from row-major f16: lane lo=row; elems 0..15 -> K=hi*16+e.
// Two aligned 16B loads.
__device__ __forceinline__ v16h load_a_frag(const _Float16* __restrict__ A, int lda,
                                            int row, int k0, int hi) {
  v8h x0 = *(const v8h*)(A + (size_t)row * lda + k0 + hi * 16);
  v8h x1 = *(const v8h*)(A + (size_t)row * lda + k0 + hi * 16 + 8);
  return __builtin_shufflevector(x0, x1, 0, 1, 2, 3, 4, 5, 6, 7,
                                 8, 9, 10, 11, 12, 13, 14, 15);
}

// B fragment (32x16 f16) from K-contiguous storage: lane holds 16 consecutive K
// values starting at k0 + hi*16, for its column. One aligned 32B load.
__device__ __forceinline__ v16h load_b_frag(const _Float16* __restrict__ BT, int ldb,
                                            int col, int k0, int hi) {
  return *(const v16h*)(BT + (size_t)col * ldb + k0 + hi * 16);
}

// ---------------- weight convert + transpose: Wt[n][k] = (f16)W[k][n] -------
__global__ void transpose_f16_kernel(const float* __restrict__ W, _Float16* __restrict__ Wt,
                                     int K, int N) {
  __shared__ float tile[32][33];
  int kb = blockIdx.x * 32, nb = blockIdx.y * 32;
  int tx = threadIdx.x, ty = threadIdx.y;  // 32 x 8
#pragma unroll
  for (int r = 0; r < 32; r += 8)
    tile[ty + r][tx] = W[(size_t)(kb + ty + r) * N + nb + tx];
  __syncthreads();
#pragma unroll
  for (int r = 0; r < 32; r += 8)
    Wt[(size_t)(nb + ty + r) * K + kb + tx] = (_Float16)tile[tx][ty + r];
}

// ---------------- embedding gather (f32 master + f16 copy) ------------------
__global__ void embed_kernel(const int* __restrict__ types,
                             const float* __restrict__ emb,
                             float* __restrict__ x, _Float16* __restrict__ xh) {
  int i = blockIdx.x, d = threadIdx.x;
  float v = emb[(size_t)types[i] * kD + d];
  x[(size_t)i * kD + d] = v;
  xh[(size_t)i * kD + d] = (_Float16)v;
}

// ---------------- f16-WMMA GEMM: C = A(MxK) @ B(KxN) (+bias, act) -----------
// A: f16 row-major. BT: f16 transposed (N x K). One wave -> 16x64 strip.
// act: 0=none, 1=exact GELU. Outputs optional: Cf f32 row-major, Ch f16
// (row-major if transC==0, column-major NxM if transC==1).
__global__ void gemm_kernel(const _Float16* __restrict__ A, const _Float16* __restrict__ BT,
                            const float* __restrict__ bias, float* __restrict__ Cf,
                            _Float16* __restrict__ Ch,
                            int M, int Kc, int Nc, int act, int transC) {
  int wave   = (int)((blockIdx.x * blockDim.x + threadIdx.x) >> 5);
  int tilesN = Nc >> 6;
  int tilesM = M >> 4;
  if (wave >= tilesM * tilesN) return;
  int row0 = (wave / tilesN) << 4;
  int col0 = (wave % tilesN) << 6;
  int lane = threadIdx.x & 31;
  int lo = lane & 15, hi = lane >> 4;

  v8f acc0 = {}, acc1 = {}, acc2 = {}, acc3 = {};
  for (int k0 = 0; k0 < Kc; k0 += 32) {
    v16h a  = load_a_frag(A, Kc, row0 + lo, k0, hi);
    v16h b0 = load_b_frag(BT, Kc, col0 +  0 + lo, k0, hi);
    v16h b1 = load_b_frag(BT, Kc, col0 + 16 + lo, k0, hi);
    v16h b2 = load_b_frag(BT, Kc, col0 + 32 + lo, k0, hi);
    v16h b3 = load_b_frag(BT, Kc, col0 + 48 + lo, k0, hi);
    acc0 = __builtin_amdgcn_wmma_f32_16x16x32_f16(false, a, false, b0, (short)0, acc0, false, false);
    acc1 = __builtin_amdgcn_wmma_f32_16x16x32_f16(false, a, false, b1, (short)0, acc1, false, false);
    acc2 = __builtin_amdgcn_wmma_f32_16x16x32_f16(false, a, false, b2, (short)0, acc2, false, false);
    acc3 = __builtin_amdgcn_wmma_f32_16x16x32_f16(false, a, false, b3, (short)0, acc3, false, false);
  }
  v8f accs[4] = {acc0, acc1, acc2, acc3};
#pragma unroll
  for (int t = 0; t < 4; ++t) {
    int col = col0 + t * 16 + lo;
    float bv = bias ? bias[col] : 0.0f;
    float vals[8];
#pragma unroll
    for (int r = 0; r < 8; ++r) {
      float v = accs[t][r] + bv;
      if (act == 1) v = 0.5f * v * (1.0f + erff(v * 0.70710678118654752f));
      vals[r] = v;
    }
    if (Cf) {
#pragma unroll
      for (int r = 0; r < 8; ++r)
        Cf[(size_t)(row0 + r + hi * 8) * Nc + col] = vals[r];
    }
    if (Ch) {
      if (transC) {
        v8h hv;
#pragma unroll
        for (int r = 0; r < 8; ++r) hv[r] = (_Float16)vals[r];
        *(v8h*)(Ch + (size_t)col * M + row0 + hi * 8) = hv;  // 16B store
      } else {
#pragma unroll
        for (int r = 0; r < 8; ++r)
          Ch[(size_t)(row0 + r + hi * 8) * Nc + col] = (_Float16)vals[r];
      }
    }
  }
}

// ---------------- flash attention with edge bias ----------------------------
// One wave per (16-row query block, head). Q,K f16 row-major (N x D);
// VT f16 column-major (D x N). Output Oh f16 row-major (N x D).
__global__ void attn_kernel(const _Float16* __restrict__ Qh, const _Float16* __restrict__ Kh,
                            const _Float16* __restrict__ VT, const int* __restrict__ eidx,
                            const float* __restrict__ eb,  // (NEDGE, H) for this layer
                            _Float16* __restrict__ Oh) {
  __shared__ __align__(32) _Float16 pbuf[16 * 32];
  int qb = blockIdx.x, h = blockIdx.y;
  int lane = threadIdx.x & 31;
  int lo = lane & 15, hi = lane >> 4;
  int row0 = qb * 16;
  const float scale = 0.17677669529663687f;  // 1/sqrt(32)

  // Q fragment: DK=32 == one WMMA K-chunk
  v16h qf = load_a_frag(Qh, kD, row0 + lo, h * kDK, hi);

  float eb0 = eb[0 * kH + h], eb1 = eb[1 * kH + h], eb2 = eb[2 * kH + h];
  float eb3 = eb[3 * kH + h], eb4 = eb[4 * kH + h];

  float mrow[8], dden[8];
#pragma unroll
  for (int r = 0; r < 8; ++r) { mrow[r] = -1e30f; dden[r] = 0.0f; }
  v8f o0 = {}, o1 = {};

  for (int k0 = 0; k0 < kN; k0 += 32) {
    // scores for 16 rows x 32 keys (two 16x16 tiles)
    v8f s[2];
#pragma unroll
    for (int t = 0; t < 2; ++t) {
      // B[dk][key]: col = key = k0+t*16+lo; 16 contiguous dk at h*32 + hi*16
      v16h kf = *(const v16h*)(Kh + (size_t)(k0 + t * 16 + lo) * kD + h * kDK + hi * 16);
      v8f z = {};
      s[t] = __builtin_amdgcn_wmma_f32_16x16x32_f16(false, qf, false, kf, (short)0, z, false, false);
    }

    // scale + edge bias, chunk row max
    float sc[2][8], cmax[8];
#pragma unroll
    for (int r = 0; r < 8; ++r) cmax[r] = -1e30f;
#pragma unroll
    for (int t = 0; t < 2; ++t)
#pragma unroll
      for (int r = 0; r < 8; ++r) {
        int row = row0 + r + hi * 8;
        int col = k0 + t * 16 + lo;
        int e = eidx[(size_t)row * kN + col];
        float bv = (e == 0) ? eb0 : (e == 1) ? eb1 : (e == 2) ? eb2 : (e == 3) ? eb3 : eb4;
        float v = s[t][r] * scale + bv;
        sc[t][r] = v;
        cmax[r] = fmaxf(cmax[r], v);
      }
#pragma unroll
    for (int m = 1; m < 16; m <<= 1)
#pragma unroll
      for (int r = 0; r < 8; ++r)
        cmax[r] = fmaxf(cmax[r], __shfl_xor(cmax[r], m, 32));

    float f[8], csum[8];
#pragma unroll
    for (int r = 0; r < 8; ++r) {
      float nm = fmaxf(mrow[r], cmax[r]);
      f[r] = __expf(mrow[r] - nm);
      mrow[r] = nm;
      csum[r] = 0.0f;
    }
    // exponentiate, write P (f16) to LDS for A-fragment re-layout
#pragma unroll
    for (int t = 0; t < 2; ++t)
#pragma unroll
      for (int r = 0; r < 8; ++r) {
        float p = __expf(sc[t][r] - mrow[r]);
        csum[r] += p;
        pbuf[(r + hi * 8) * 32 + t * 16 + lo] = (_Float16)p;
      }
#pragma unroll
    for (int m = 1; m < 16; m <<= 1)
#pragma unroll
      for (int r = 0; r < 8; ++r)
        csum[r] += __shfl_xor(csum[r], m, 32);
#pragma unroll
    for (int r = 0; r < 8; ++r) {
      dden[r] = dden[r] * f[r] + csum[r];
      o0[r] *= f[r];
      o1[r] *= f[r];
    }
    __syncthreads();
    v16h pf = load_a_frag(pbuf, 32, lo, 0, hi);  // P as 16x32 A fragment
    __syncthreads();

    // O += P @ V  (two dk-halves); VT gives contiguous keys per dk
#pragma unroll
    for (int t = 0; t < 2; ++t) {
      v16h vf = *(const v16h*)(VT + (size_t)(h * kDK + t * 16 + lo) * kN + k0 + hi * 16);
      if (t == 0)
        o0 = __builtin_amdgcn_wmma_f32_16x16x32_f16(false, pf, false, vf, (short)0, o0, false, false);
      else
        o1 = __builtin_amdgcn_wmma_f32_16x16x32_f16(false, pf, false, vf, (short)0, o1, false, false);
    }
  }
#pragma unroll
  for (int r = 0; r < 8; ++r) {
    float inv = 1.0f / dden[r];
    size_t row = (size_t)(row0 + r + hi * 8);
    Oh[row * kD + h * kDK + lo]      = (_Float16)(o0[r] * inv);
    Oh[row * kD + h * kDK + 16 + lo] = (_Float16)(o1[r] * inv);
  }
}

// ---------------- residual add + LayerNorm (one block per row) --------------
__global__ void add_ln_kernel(const float* __restrict__ xin, const float* __restrict__ y,
                              const float* __restrict__ g, const float* __restrict__ b,
                              float* __restrict__ xout, _Float16* __restrict__ xh) {
  __shared__ float red[kD];
  int row = blockIdx.x, d = threadIdx.x;
  float v = xin[(size_t)row * kD + d] + y[(size_t)row * kD + d];
  red[d] = v;
  __syncthreads();
  for (int s = kD / 2; s > 0; s >>= 1) { if (d < s) red[d] += red[d + s]; __syncthreads(); }
  float mean = red[0] * (1.0f / kD);
  __syncthreads();
  float dv = v - mean;
  red[d] = dv * dv;
  __syncthreads();
  for (int s = kD / 2; s > 0; s >>= 1) { if (d < s) red[d] += red[d + s]; __syncthreads(); }
  float var = red[0] * (1.0f / kD);
  float o = dv * rsqrtf(var + 1e-5f) * g[d] + b[d];
  xout[(size_t)row * kD + d] = o;
  xh[(size_t)row * kD + d] = (_Float16)o;
}

// ---------------- pooling (mean + max over rows) ----------------------------
__global__ void pool_kernel(const float* __restrict__ x, float* __restrict__ pooled) {
  int d = threadIdx.x;
  float s = 0.0f, mx = -1e30f;
  for (int i = 0; i < kN; ++i) {
    float v = x[(size_t)i * kD + d];
    s += v;
    mx = fmaxf(mx, v);
  }
  pooled[d] = s * (1.0f / kN);
  pooled[kD + d] = mx;
}

// ---------------- final projection: out = pooled(512) @ Wr(512x256) + br ----
__global__ void final_kernel(const float* __restrict__ pooled, const float* __restrict__ Wr,
                             const float* __restrict__ br, float* __restrict__ out) {
  int j = threadIdx.x;
  float acc = br[j];
  for (int k = 0; k < 2 * kD; ++k) acc += pooled[k] * Wr[(size_t)k * kD + j];
  out[j] = acc;
}

extern "C" void kernel_launch(void* const* d_in, const int* in_sizes, int n_in,
                              void* d_out, int out_size, void* d_ws, size_t ws_size,
                              hipStream_t stream) {
  const int*   node_types = (const int*)d_in[0];
  const int*   eidx       = (const int*)d_in[1];
  const float* node_emb   = (const float*)d_in[2];
  const float* Wq  = (const float*)d_in[3];
  const float* Wk  = (const float*)d_in[4];
  const float* Wv  = (const float*)d_in[5];
  const float* Wo  = (const float*)d_in[6];
  const float* bo  = (const float*)d_in[7];
  const float* eb  = (const float*)d_in[8];
  const float* W1  = (const float*)d_in[9];
  const float* b1  = (const float*)d_in[10];
  const float* W2  = (const float*)d_in[11];
  const float* b2  = (const float*)d_in[12];
  const float* g1  = (const float*)d_in[13];
  const float* be1 = (const float*)d_in[14];
  const float* g2  = (const float*)d_in[15];
  const float* be2 = (const float*)d_in[16];
  const float* Wr  = (const float*)d_in[17];
  const float* br  = (const float*)d_in[18];
  float* out = (float*)d_out;

  // ---- workspace carve-up (all regions >= 32B aligned) ----
  char* base = (char*)d_ws;
  auto alloc_f = [&](size_t n) { float* p = (float*)base; base += n * sizeof(float); return p; };
  auto alloc_h = [&](size_t n) { _Float16* p = (_Float16*)base; base += n * sizeof(_Float16); return p; };

  float* x      = alloc_f((size_t)kN * kD);
  float* y      = alloc_f((size_t)kN * kD);
  float* pooled = alloc_f(2 * kD);
  _Float16* xh   = alloc_h((size_t)kN * kD);
  _Float16* qh   = alloc_h((size_t)kN * kD);
  _Float16* kh   = alloc_h((size_t)kN * kD);
  _Float16* vth  = alloc_h((size_t)kD * kN);   // transposed V (D x N)
  _Float16* atth = alloc_h((size_t)kN * kD);
  _Float16* hh   = alloc_h((size_t)kN * kDFF);
  _Float16* WqT  = alloc_h((size_t)kL * kD * kD);
  _Float16* WkT  = alloc_h((size_t)kL * kD * kD);
  _Float16* WvT  = alloc_h((size_t)kL * kD * kD);
  _Float16* WoT  = alloc_h((size_t)kL * kD * kD);
  _Float16* W1T  = alloc_h((size_t)kL * kD * kDFF);  // (DFF x D) per layer
  _Float16* W2T  = alloc_h((size_t)kL * kDFF * kD);  // (D x DFF) per layer

  // ---- one-time weight convert+transpose to f16 ----
  dim3 tb(32, 8);
  for (int l = 0; l < kL; ++l) {
    transpose_f16_kernel<<<dim3(kD / 32, kD / 32), tb, 0, stream>>>(
        Wq + (size_t)l * kD * kD, WqT + (size_t)l * kD * kD, kD, kD);
    transpose_f16_kernel<<<dim3(kD / 32, kD / 32), tb, 0, stream>>>(
        Wk + (size_t)l * kD * kD, WkT + (size_t)l * kD * kD, kD, kD);
    transpose_f16_kernel<<<dim3(kD / 32, kD / 32), tb, 0, stream>>>(
        Wv + (size_t)l * kD * kD, WvT + (size_t)l * kD * kD, kD, kD);
    transpose_f16_kernel<<<dim3(kD / 32, kD / 32), tb, 0, stream>>>(
        Wo + (size_t)l * kD * kD, WoT + (size_t)l * kD * kD, kD, kD);
    transpose_f16_kernel<<<dim3(kD / 32, kDFF / 32), tb, 0, stream>>>(
        W1 + (size_t)l * kD * kDFF, W1T + (size_t)l * kD * kDFF, kD, kDFF);
    transpose_f16_kernel<<<dim3(kDFF / 32, kD / 32), tb, 0, stream>>>(
        W2 + (size_t)l * kDFF * kD, W2T + (size_t)l * kDFF * kD, kDFF, kD);
  }

  embed_kernel<<<kN, kD, 0, stream>>>(node_types, node_emb, x, xh);

  const int blkDD = ((kN / 16) * (kD / 64)) / 8;    // 2048x256x256 GEMMs
  const int blkDF = ((kN / 16) * (kDFF / 64)) / 8;  // 2048x256x1024 GEMM

  for (int l = 0; l < kL; ++l) {
    const float* bo_  = bo + (size_t)l * kD;
    const float* ebl  = eb + (size_t)l * 5 * kH;
    const float* b1_  = b1 + (size_t)l * kDFF;
    const float* b2_  = b2 + (size_t)l * kD;
    const _Float16* wqT = WqT + (size_t)l * kD * kD;
    const _Float16* wkT = WkT + (size_t)l * kD * kD;
    const _Float16* wvT = WvT + (size_t)l * kD * kD;
    const _Float16* woT = WoT + (size_t)l * kD * kD;
    const _Float16* w1T = W1T + (size_t)l * kD * kDFF;
    const _Float16* w2T = W2T + (size_t)l * kDFF * kD;

    gemm_kernel<<<blkDD, 256, 0, stream>>>(xh, wqT, nullptr, nullptr, qh,  kN, kD, kD, 0, 0);
    gemm_kernel<<<blkDD, 256, 0, stream>>>(xh, wkT, nullptr, nullptr, kh,  kN, kD, kD, 0, 0);
    gemm_kernel<<<blkDD, 256, 0, stream>>>(xh, wvT, nullptr, nullptr, vth, kN, kD, kD, 0, 1);

    attn_kernel<<<dim3(kN / 16, kH), 32, 0, stream>>>(qh, kh, vth, eidx, ebl, atth);

    gemm_kernel<<<blkDD, 256, 0, stream>>>(atth, woT, bo_, y, nullptr, kN, kD, kD, 0, 0);
    add_ln_kernel<<<kN, kD, 0, stream>>>(x, y, g1 + (size_t)l * kD, be1 + (size_t)l * kD, x, xh);

    gemm_kernel<<<blkDF, 256, 0, stream>>>(xh, w1T, b1_, nullptr, hh, kN, kD, kDFF, 1, 0);
    gemm_kernel<<<blkDD, 256, 0, stream>>>(hh, w2T, b2_, y, nullptr, kN, kDFF, kD, 0, 0);
    add_ln_kernel<<<kN, kD, 0, stream>>>(x, y, g2 + (size_t)l * kD, be2 + (size_t)l * kD, x, xh);
  }

  pool_kernel<<<1, kD, 0, stream>>>(x, pooled);
  final_kernel<<<1, kD, 0, stream>>>(pooled, Wr, br, out);
}
